// HippoModel_23785528885716
// MI455X (gfx1250) — matrix operand order repaired and checked by reading, not compiled, not run on hardware
//
#include <hip/hip_runtime.h>
#include <hip/hip_bf16.h>
#include <math.h>

// ---------------------------------------------------------------------------
// HiPPO block for MI455X (gfx1250, wave32, WMMA).
//  - All heavy GEMMs: v_wmma_f32_16x16x32_bf16, fp32 accumulate.
//  - Fused C-projection: ssm[m,o] = sum_h hs[m,h]*(P_h)*sigmoid(G_h) computed
//    tile-wise in registers; the 2x256MB C/G tensors never touch HBM.
//  - Weights/activations pre-converted fp32->bf16 once (halves streamed bytes;
//    Wc+Wcg bf16 = 64MB fits the 192MB L2, so cross-block panel re-reads hit L2).
//  - Tile staging uses GLOBAL_LOAD_ASYNC_TO_LDS_* (ASYNCcnt + s_wait_asynccnt).
// ---------------------------------------------------------------------------

#define DD   1024
#define HH   16
#define OO   1024
#define FF   4096
#define BSZ  4
#define LL   1024
#define ROWS (BSZ * LL)   // 4096

typedef __attribute__((ext_vector_type(16))) __bf16 v16bf;
typedef __attribute__((ext_vector_type(8)))  float  v8f;

// GCC-style vector types matching the async builtin parameter types
typedef int a_v4i __attribute__((vector_size(16)));
typedef int a_v2i __attribute__((vector_size(8)));
typedef __attribute__((address_space(1))) a_v4i* g_v4i_p;
typedef __attribute__((address_space(3))) a_v4i* l_v4i_p;
typedef __attribute__((address_space(1))) a_v2i* g_v2i_p;
typedef __attribute__((address_space(3))) a_v2i* l_v2i_p;

// ---------------- feature detection -----------------------------------------
#if defined(__has_builtin)
#  if __has_builtin(__builtin_amdgcn_global_load_async_to_lds_b128) && \
      __has_builtin(__builtin_amdgcn_global_load_async_to_lds_b64)
#    define HAS_ASYNC_LDS 1
#  endif
#  if __has_builtin(__builtin_amdgcn_s_wait_asynccnt)
#    define HAS_WAIT_ASYNC 1
#  endif
#  if __has_builtin(__builtin_amdgcn_cvt_pk_bf16_f32)
#    define HAS_PK_BF16 1
#  endif
#endif
#ifndef HAS_ASYNC_LDS
#  define HAS_ASYNC_LDS 0
#endif
#ifndef HAS_WAIT_ASYNC
#  define HAS_WAIT_ASYNC 0
#endif
#ifndef HAS_PK_BF16
#  define HAS_PK_BF16 0
#endif

// ---------------- helpers ----------------------------------------------------
__device__ __forceinline__ unsigned short f2bf(float f) {
    unsigned int u = __builtin_bit_cast(unsigned int, f);
    unsigned int r = u + 0x7FFFu + ((u >> 16) & 1u);   // round-to-nearest-even
    return (unsigned short)(r >> 16);
}

__device__ __forceinline__ unsigned int f2bf2(float x, float y) {
#if HAS_PK_BF16
    typedef __attribute__((ext_vector_type(2))) __bf16 v2bf;
    v2bf r = __builtin_amdgcn_cvt_pk_bf16_f32(x, y);   // v_cvt_pk_bf16_f32
    return __builtin_bit_cast(unsigned int, r);
#else
    return (unsigned int)f2bf(x) | ((unsigned int)f2bf(y) << 16);
#endif
}

__device__ __forceinline__ v8f zero8() {
    v8f z = {0.f, 0.f, 0.f, 0.f, 0.f, 0.f, 0.f, 0.f};
    return z;
}

// async (or fallback) global-bf16 -> LDS tile copies
__device__ __forceinline__ void cp_g2l_b128(unsigned short* lds, const unsigned short* g) {
#if HAS_ASYNC_LDS
    __builtin_amdgcn_global_load_async_to_lds_b128((g_v4i_p)(g), (l_v4i_p)(lds), 0, 0);
#else
    *(uint4*)lds = *(const uint4*)g;
#endif
}
__device__ __forceinline__ void cp_g2l_b64(unsigned short* lds, const unsigned short* g) {
#if HAS_ASYNC_LDS
    __builtin_amdgcn_global_load_async_to_lds_b64((g_v2i_p)(g), (l_v2i_p)(lds), 0, 0);
#else
    *(uint2*)lds = *(const uint2*)g;
#endif
}
__device__ __forceinline__ void wait_async_all() {
#if HAS_ASYNC_LDS && HAS_WAIT_ASYNC
    __builtin_amdgcn_s_wait_asynccnt(0);               // s_wait_asynccnt 0
#endif
}

// A fragment (16x32 bf16, ISA 7.12.2): lane m=lane&15, hi=lane>>4;
// a[0..7] = K[hi*8 .. +7], a[8..15] = K[16+hi*8 .. +7]
__device__ __forceinline__ v16bf load_frag_A(const unsigned short* lds, int mBase, int lane) {
    int m  = mBase + (lane & 15);
    int hi = lane >> 4;
    const unsigned short* p = lds + m * 32 + hi * 8;
    union { uint4 u[2]; v16bf v; } cv;
    cv.u[0] = *(const uint4*)(p);
    cv.u[1] = *(const uint4*)(p + 16);
    return cv.v;
}

// B fragment (32x16 bf16): lane n=lane&15, hi=lane>>4; b[0..7] = K[hi*8 .. +7], b[8..15] = K[16+hi*8 .. +7]
__device__ __forceinline__ v16bf load_frag_B(const unsigned short* lds, int nBase, int lane) {
    int n  = nBase + (lane & 15);
    int hi = lane >> 4;
    const unsigned short* p = lds + n * 32 + hi * 8;
    union { uint4 u[2]; v16bf v; } cv;
    cv.u[0] = *(const uint4*)(p);
    cv.u[1] = *(const uint4*)(p + 16);
    return cv.v;
}

__device__ __forceinline__ v8f wmma_bf16(v16bf a, v16bf b, v8f c) {
    return __builtin_amdgcn_wmma_f32_16x16x32_bf16(false, a, false, b, (short)0, c, false, false);
}

__device__ __forceinline__ float sigmoidf_(float x) { return 1.f / (1.f + __expf(-x)); }

// ---------------------------------------------------------------------------
// fp32 -> bf16 bulk conversion (n must be a multiple of 4)
// ---------------------------------------------------------------------------
__global__ __launch_bounds__(256) void k_cvt(const float* __restrict__ in,
                                             unsigned short* __restrict__ out, int n4) {
    int i = blockIdx.x * 256 + threadIdx.x;
    if (i < n4) {
        float4 f = ((const float4*)in)[i];
        uint2 u;
        u.x = f2bf2(f.x, f.y);
        u.y = f2bf2(f.z, f.w);
        ((uint2*)out)[i] = u;
    }
}

// ---------------------------------------------------------------------------
// LayerNorm over last dim (1024). One block per row.
// ---------------------------------------------------------------------------
__global__ __launch_bounds__(256) void k_layernorm(const float* __restrict__ in,
                                                   const float* __restrict__ g,
                                                   const float* __restrict__ b,
                                                   float* __restrict__ out) {
    __shared__ float red[8];
    size_t row = blockIdx.x;
    const float* x = in + row * DD;
    int tid = threadIdx.x, lane = tid & 31, wid = tid >> 5;

    float s = 0.f;
    for (int i = tid; i < DD; i += 256) s += x[i];
    for (int o = 16; o; o >>= 1) s += __shfl_xor(s, o, 32);
    if (lane == 0) red[wid] = s;
    __syncthreads();
    float tot = 0.f;
    for (int i = 0; i < 8; ++i) tot += red[i];
    float mean = tot * (1.f / DD);
    __syncthreads();

    float vs = 0.f;
    for (int i = tid; i < DD; i += 256) { float d = x[i] - mean; vs += d * d; }
    for (int o = 16; o; o >>= 1) vs += __shfl_xor(vs, o, 32);
    if (lane == 0) red[wid] = vs;
    __syncthreads();
    float vtot = 0.f;
    for (int i = 0; i < 8; ++i) vtot += red[i];
    float inv = rsqrtf(vtot * (1.f / DD) + 1e-5f);

    for (int i = tid; i < DD; i += 256)
        out[row * DD + i] = (x[i] - mean) * inv * g[i] + b[i];
}

// ---------------------------------------------------------------------------
// Gated B projection: B = (xn@Wb.T + bb) * sigmoid(xn@Wbg.T + bbg). H=16.
// ---------------------------------------------------------------------------
__global__ __launch_bounds__(256) void k_projB(const float* __restrict__ xn,
                                               const float* __restrict__ Wb,  const float* __restrict__ bb,
                                               const float* __restrict__ Wbg, const float* __restrict__ bbg,
                                               float* __restrict__ Bout) {
    __shared__ float partial[32];
    size_t row = blockIdx.x;
    const float* x = xn + row * DD;
    int lane = threadIdx.x & 31, wid = threadIdx.x >> 5;

    for (int j = wid; j < 32; j += 8) {
        int h = j & 15;
        const float* w = ((j >= 16) ? Wbg : Wb) + (size_t)h * DD;
        float s = 0.f;
        for (int k = lane; k < DD; k += 32) s += x[k] * w[k];
        for (int o = 16; o; o >>= 1) s += __shfl_xor(s, o, 32);
        if (lane == 0) partial[j] = s;
    }
    __syncthreads();
    if (threadIdx.x < 16) {
        int h = threadIdx.x;
        float p  = partial[h] + bb[h];
        float gv = partial[16 + h] + bbg[h];
        Bout[row * HH + h] = p * sigmoidf_(gv);
    }
}

// ---------------------------------------------------------------------------
// Sequential selective scan. last_h == final state (== reference decay-sum).
// ---------------------------------------------------------------------------
__global__ void k_scan(const float* __restrict__ Bout, const float* __restrict__ A,
                       const float* __restrict__ h0, float* __restrict__ hs,
                       float* __restrict__ lastH) {
    int t = threadIdx.x;
    if (t >= BSZ * HH) return;
    int b = t >> 4, h = t & 15;
    float d  = __expf(A[h]);
    float hv = h0[b * HH + h];
    for (int i = 0; i < LL; ++i) {
        size_t idx = ((size_t)b * LL + i) * HH + h;
        hv = d * hv + Bout[idx];
        hs[idx] = hv;
    }
    lastH[b * HH + h] = hv;
}

// ---------------------------------------------------------------------------
// Fused SSM: y1 = bv + bd + xn@Wd.T + sum_h hs[:,h]*(xn@Wc_h.T+bc_h)*sig(xn@Wcg_h.T+bcg_h)
// Macro tile 32(M) x 64(N); 8 waves = 2x4 grid of 16x16 WMMA tiles.
// ---------------------------------------------------------------------------
__global__ __launch_bounds__(256) void k_ssm(const unsigned short* __restrict__ xnb,
                                             const float* __restrict__ bv,
                                             const float* __restrict__ hs,
                                             const unsigned short* __restrict__ Wcb,
                                             const float* __restrict__ bc,
                                             const unsigned short* __restrict__ Wcgb,
                                             const float* __restrict__ bcg,
                                             const unsigned short* __restrict__ Wdb,
                                             const float* __restrict__ bd,
                                             float* __restrict__ y1) {
    __shared__ unsigned short At[32 * 32];
    __shared__ unsigned short Bt[2][64 * 32];
    __shared__ float hsT[32 * 16];

    int tid = threadIdx.x, lane = tid & 31, wid = tid >> 5;
    int wave_m = wid >> 2, wave_n = wid & 3;          // 2 x 4 waves
    int blockM = blockIdx.y * 32;
    int blockN = blockIdx.x * 64;

    for (int i = tid; i < 32 * 16; i += 256)
        hsT[i] = hs[(size_t)(blockM + (i >> 4)) * HH + (i & 15)];

    int hi = lane >> 4, nl = lane & 15;
    int n_g = blockN + wave_n * 16 + nl;
    int ar = tid >> 3, ac = (tid & 7) * 4;            // A-tile: 8B per thread
    int br = tid >> 2, bcol = (tid & 3) * 8;          // B-tile: 16B per thread

    v8f accS = zero8();
    for (int h = 0; h < HH; ++h) {
        v8f accP = zero8(), accG = zero8();
        for (int kt = 0; kt < DD; kt += 32) {
            __syncthreads();                            // LDS free to overwrite
            cp_g2l_b64(&At[ar * 32 + ac], xnb + (size_t)(blockM + ar) * DD + kt + ac);
            size_t wrow = ((size_t)h * OO + blockN + br) * DD + kt + bcol;
            cp_g2l_b128(&Bt[0][br * 32 + bcol], Wcb  + wrow);
            cp_g2l_b128(&Bt[1][br * 32 + bcol], Wcgb + wrow);
            wait_async_all();
            __syncthreads();
            v16bf a  = load_frag_A(At, wave_m * 16, lane);
            v16bf b0 = load_frag_B(Bt[0], wave_n * 16, lane);
            v16bf b1 = load_frag_B(Bt[1], wave_n * 16, lane);
            accP = wmma_bf16(a, b0, accP);
            accG = wmma_bf16(a, b1, accG);
        }
        // gating + h-contraction epilogue (C never hits memory)
        float bcv  = bc [(size_t)h * OO + n_g];
        float bcgv = bcg[(size_t)h * OO + n_g];
        for (int r = 0; r < 8; ++r) {
            int m_loc = wave_m * 16 + hi * 8 + r;
            float P = accP[r] + bcv;
            float G = accG[r] + bcgv;
            accS[r] += hsT[m_loc * 16 + h] * P * sigmoidf_(G);
        }
    }

    // Wd projection (Dout) with the same tiling
    v8f accD = zero8();
    for (int kt = 0; kt < DD; kt += 32) {
        __syncthreads();
        cp_g2l_b64(&At[ar * 32 + ac], xnb + (size_t)(blockM + ar) * DD + kt + ac);
        cp_g2l_b128(&Bt[0][br * 32 + bcol], Wdb + (size_t)(blockN + br) * DD + kt + bcol);
        wait_async_all();
        __syncthreads();
        v16bf a  = load_frag_A(At, wave_m * 16, lane);
        v16bf b0 = load_frag_B(Bt[0], wave_n * 16, lane);
        accD = wmma_bf16(a, b0, accD);
    }

    float bdv = bd[n_g];
    for (int r = 0; r < 8; ++r) {
        size_t m_g = blockM + wave_m * 16 + hi * 8 + r;
        y1[m_g * OO + n_g] = bv[m_g * OO + n_g] + accS[r] + accD[r] + bdv;
    }
}

// ---------------------------------------------------------------------------
// FFN1: Hg = gelu(y2 @ W1.T + b1), exact erf GeLU, bf16 output.
// ---------------------------------------------------------------------------
__global__ __launch_bounds__(256) void k_ffn1(const unsigned short* __restrict__ y2b,
                                              const unsigned short* __restrict__ W1b,
                                              const float* __restrict__ b1,
                                              unsigned short* __restrict__ Hg) {
    __shared__ unsigned short At[32 * 32];
    __shared__ unsigned short Bt[64 * 32];
    int tid = threadIdx.x, lane = tid & 31, wid = tid >> 5;
    int wave_m = wid >> 2, wave_n = wid & 3;
    int blockM = blockIdx.y * 32;
    int blockN = blockIdx.x * 64;
    int ar = tid >> 3, ac = (tid & 7) * 4;
    int br = tid >> 2, bcol = (tid & 3) * 8;

    v8f acc = zero8();
    for (int kt = 0; kt < DD; kt += 32) {
        __syncthreads();
        cp_g2l_b64(&At[ar * 32 + ac], y2b + (size_t)(blockM + ar) * DD + kt + ac);
        cp_g2l_b128(&Bt[br * 32 + bcol], W1b + (size_t)(blockN + br) * DD + kt + bcol);
        wait_async_all();
        __syncthreads();
        v16bf a = load_frag_A(At, wave_m * 16, lane);
        v16bf b = load_frag_B(Bt, wave_n * 16, lane);
        acc = wmma_bf16(a, b, acc);
    }

    int hi = lane >> 4, nl = lane & 15;
    int n_g = blockN + wave_n * 16 + nl;
    float b1v = b1[n_g];
    for (int r = 0; r < 8; ++r) {
        size_t m_g = blockM + wave_m * 16 + hi * 8 + r;
        float v = acc[r] + b1v;
        v = 0.5f * v * (1.f + erff(v * 0.70710678118654752f));   // exact gelu
        Hg[m_g * FF + n_g] = f2bf(v);
    }
}

// ---------------------------------------------------------------------------
// FFN2: out = y2 + Hg @ W2.T + b2.
// ---------------------------------------------------------------------------
__global__ __launch_bounds__(256) void k_ffn2(const unsigned short* __restrict__ Hg,
                                              const unsigned short* __restrict__ W2b,
                                              const float* __restrict__ b2,
                                              const float* __restrict__ y2,
                                              float* __restrict__ out) {
    __shared__ unsigned short At[32 * 32];
    __shared__ unsigned short Bt[64 * 32];
    int tid = threadIdx.x, lane = tid & 31, wid = tid >> 5;
    int wave_m = wid >> 2, wave_n = wid & 3;
    int blockM = blockIdx.y * 32;
    int blockN = blockIdx.x * 64;
    int ar = tid >> 3, ac = (tid & 7) * 4;
    int br = tid >> 2, bcol = (tid & 3) * 8;

    v8f acc = zero8();
    for (int kt = 0; kt < FF; kt += 32) {
        __syncthreads();
        cp_g2l_b64(&At[ar * 32 + ac], Hg + (size_t)(blockM + ar) * FF + kt + ac);
        cp_g2l_b128(&Bt[br * 32 + bcol], W2b + (size_t)(blockN + br) * FF + kt + bcol);
        wait_async_all();
        __syncthreads();
        v16bf a = load_frag_A(At, wave_m * 16, lane);
        v16bf b = load_frag_B(Bt, wave_n * 16, lane);
        acc = wmma_bf16(a, b, acc);
    }

    int hi = lane >> 4, nl = lane & 15;
    int n_g = blockN + wave_n * 16 + nl;
    float b2v = b2[n_g];
    for (int r = 0; r < 8; ++r) {
        size_t m_g = blockM + wave_m * 16 + hi * 8 + r;
        out[m_g * DD + n_g] = y2[m_g * DD + n_g] + acc[r] + b2v;
    }
}

// ---------------------------------------------------------------------------
extern "C" void kernel_launch(void* const* d_in, const int* in_sizes, int n_in,
                              void* d_out, int out_size, void* d_ws, size_t ws_size,
                              hipStream_t stream) {
    const float* bv  = (const float*)d_in[0];
    const float* h0  = (const float*)d_in[1];
    const float* A   = (const float*)d_in[2];
    const float* Wb  = (const float*)d_in[3];
    const float* bb  = (const float*)d_in[4];
    const float* Wc  = (const float*)d_in[5];
    const float* bc  = (const float*)d_in[6];
    const float* Wd  = (const float*)d_in[7];
    const float* bd  = (const float*)d_in[8];
    const float* Wbg = (const float*)d_in[9];
    const float* bbg = (const float*)d_in[10];
    const float* Wcg = (const float*)d_in[11];
    const float* bcg = (const float*)d_in[12];
    const float* W1  = (const float*)d_in[13];
    const float* b1  = (const float*)d_in[14];
    const float* W2  = (const float*)d_in[15];
    const float* b2  = (const float*)d_in[16];
    const float* g1  = (const float*)d_in[17];
    const float* be1 = (const float*)d_in[18];
    const float* g2  = (const float*)d_in[19];
    const float* be2 = (const float*)d_in[20];

    // ---- workspace layout (~187 MB) ----
    char* ws = (char*)d_ws;
    size_t off = 0;
    float* xn  = (float*)(ws + off); off += (size_t)ROWS * DD * 4;       // 16 MB
    float* y1  = (float*)(ws + off); off += (size_t)ROWS * DD * 4;       // 16 MB
    float* y2  = (float*)(ws + off); off += (size_t)ROWS * DD * 4;       // 16 MB
    float* Bp  = (float*)(ws + off); off += (size_t)ROWS * HH * 4;       // 256 KB
    float* hsv = (float*)(ws + off); off += (size_t)ROWS * HH * 4;       // 256 KB
    unsigned short* Hg   = (unsigned short*)(ws + off); off += (size_t)ROWS * FF * 2;   // 32 MB
    unsigned short* xnb  = (unsigned short*)(ws + off); off += (size_t)ROWS * DD * 2;   // 8 MB
    unsigned short* y2b  = (unsigned short*)(ws + off); off += (size_t)ROWS * DD * 2;   // 8 MB
    unsigned short* Wcb  = (unsigned short*)(ws + off); off += (size_t)HH * OO * DD * 2; // 33.5 MB
    unsigned short* Wcgb = (unsigned short*)(ws + off); off += (size_t)HH * OO * DD * 2; // 33.5 MB
    unsigned short* Wdb  = (unsigned short*)(ws + off); off += (size_t)OO * DD * 2;      // 2 MB
    unsigned short* W1b  = (unsigned short*)(ws + off); off += (size_t)FF * DD * 2;      // 8 MB
    unsigned short* W2b  = (unsigned short*)(ws + off); off += (size_t)DD * FF * 2;      // 8 MB

    float* out   = (float*)d_out;
    float* lastH = out + (size_t)ROWS * DD;

    // weight bf16 mirrors (one pass; deterministic every call)
    auto cvt = [&](const float* src, unsigned short* dst, size_t n) {
        int n4 = (int)(n / 4);
        k_cvt<<<(n4 + 255) / 256, 256, 0, stream>>>(src, dst, n4);
    };
    cvt(Wc,  Wcb,  (size_t)HH * OO * DD);
    cvt(Wcg, Wcgb, (size_t)HH * OO * DD);
    cvt(Wd,  Wdb,  (size_t)OO * DD);
    cvt(W1,  W1b,  (size_t)FF * DD);
    cvt(W2,  W2b,  (size_t)DD * FF);

    k_layernorm<<<ROWS, 256, 0, stream>>>(bv, g1, be1, xn);
    cvt(xn, xnb, (size_t)ROWS * DD);
    k_projB<<<ROWS, 256, 0, stream>>>(xn, Wb, bb, Wbg, bbg, Bp);
    k_scan<<<1, 64, 0, stream>>>(Bp, A, h0, hsv, lastH);
    k_ssm<<<dim3(OO / 64, ROWS / 32), 256, 0, stream>>>(xnb, bv, hsv, Wcb, bc, Wcgb, bcg, Wdb, bd, y1);
    k_layernorm<<<ROWS, 256, 0, stream>>>(y1, g2, be2, y2);
    cvt(y2, y2b, (size_t)ROWS * DD);
    k_ffn1<<<dim3(FF / 64, ROWS / 32), 256, 0, stream>>>(y2b, W1b, b1, Hg);
    k_ffn2<<<dim3(DD / 64, ROWS / 32), 256, 0, stream>>>(Hg, W2b, b2, y2, out);
}
